// RWKV_v7_TimeMix_26242250178958
// MI455X (gfx1250) — matrix operand order repaired and checked by reading, not compiled, not run on hardware
//
#include <hip/hip_runtime.h>
#include <cstdint>
#include <cstddef>

// ---------------- problem constants ----------------
#define Bq 4
#define Tq 1024
#define Cq 1024
#define Hq 16
#define Nq 64
#define Mq (Bq*Tq)          // 4096 rows
#define EPSgn 0.00064f

typedef unsigned short u16;
typedef __attribute__((ext_vector_type(16))) __bf16 bf16x16;
typedef __attribute__((ext_vector_type(8)))  float  f32x8;
typedef __attribute__((ext_vector_type(4)))  float  f32x4;

union FragBF { bf16x16 bf; f32x4 q[2]; };

__device__ __forceinline__ u16 f32_to_bf16(float f) {
  union { float f; unsigned int u; } v; v.f = f;
  unsigned int r = v.u + (0x7FFFu + ((v.u >> 16) & 1u));   // round-nearest-even
  return (u16)(r >> 16);
}
__device__ __forceinline__ float sigf(float x) { return 1.0f / (1.0f + expf(-x)); }

// ---------------- f32 -> bf16 weight conversion ----------------
__global__ __launch_bounds__(256) void conv_f32_bf16(const float* __restrict__ src,
                                                     u16* __restrict__ dst, int n) {
  int i = blockIdx.x * 256 + threadIdx.x;
  if (i < n) dst[i] = f32_to_bf16(src[i]);
}

// ---------------- time-shift + token mixing (6 streams, bf16 out) ----------------
__global__ __launch_bounds__(256) void prep_mix(
    const float* __restrict__ x,
    const float* __restrict__ pr, const float* __restrict__ pw,
    const float* __restrict__ pk, const float* __restrict__ pv,
    const float* __restrict__ pa, const float* __restrict__ pg,
    u16* __restrict__ mr, u16* __restrict__ mw, u16* __restrict__ mk,
    u16* __restrict__ mv, u16* __restrict__ ma, u16* __restrict__ mg) {
  size_t i = (size_t)blockIdx.x * 256 + threadIdx.x;
  int c = (int)(i & (Cq - 1));
  int t = (int)((i / Cq) & (Tq - 1));
  float xc = x[i];
  float xp = (t == 0) ? 0.0f : x[i - Cq];
  float xx = xp - xc;                     // shift(x) - x
  mr[i] = f32_to_bf16(fmaf(xx, pr[c], xc));
  mw[i] = f32_to_bf16(fmaf(xx, pw[c], xc));
  mk[i] = f32_to_bf16(fmaf(xx, pk[c], xc));
  mv[i] = f32_to_bf16(fmaf(xx, pv[c], xc));
  ma[i] = f32_to_bf16(fmaf(xx, pa[c], xc));
  mg[i] = f32_to_bf16(fmaf(xx, pg[c], xc));
}

// ---------------- bf16 WMMA GEMM: out[M,N] = A[M,K] * W[N,K]^T, fused epilogue ----
// Double-buffered LDS, BK=64 (2 WMMA k-steps per stage), one barrier per stage.
// MODE 0: plain f32 store
// MODE 1: decay  w = exp(max(-softplus(-(w0 + tanh(v)*w2)) - 0.5, -10))
// MODE 2: a = sigmoid(a0 + v*a2)
// MODE 3: g = sigmoid(v)*g2
#define BM 128
#define BN 128
#define BK 64
#define LDT 72   // padded LDS row stride (bf16): 144B -> bank stride 36, distinct over 16 rows

template <int MODE>
__global__ __launch_bounds__(256) void gemm_bf16_wmma(
    const u16* __restrict__ A, const u16* __restrict__ W, float* __restrict__ out,
    const float* __restrict__ p0, const float* __restrict__ p1,
    int Mdim, int Ndim, int Kdim) {
  __shared__ u16 sA[2][BM][LDT];   // 2 x 18 KB
  __shared__ u16 sB[2][BN][LDT];   // 2 x 18 KB  (72 KB total)
  const int tid  = threadIdx.x;
  const int lane = tid & 31;
  const int wave = tid >> 5;
  const int wm   = wave >> 2;      // 0..1 : 64-row slab
  const int wn   = wave & 3;       // 0..3 : 32-col slab
  const int hi   = lane >> 4;      // half-wave
  const int l16  = lane & 15;
  const int bm   = blockIdx.y * BM;
  const int bn   = blockIdx.x * BN;

  f32x8 acc[4][2] = {};

  // staging: thread covers row r, 32 k-elems at seg*32
  const int r   = tid >> 1;        // 0..127
  const int seg = tid & 1;         // 0..1
  const u16* gA = A + (size_t)(bm + r) * Kdim + seg * 32;
  const u16* gB = W + (size_t)(bn + r) * Kdim + seg * 32;
  const int nk = Kdim / BK;

  f32x4 ra[4], rb[4];
  // preload stage 0 into registers
#pragma unroll
  for (int q = 0; q < 4; ++q) {
    ra[q] = *(const f32x4*)(gA + q * 8);
    rb[q] = *(const f32x4*)(gB + q * 8);
  }
  // deposit stage 0
#pragma unroll
  for (int q = 0; q < 4; ++q) {
    *(f32x4*)&sA[0][r][seg * 32 + q * 8] = ra[q];
    *(f32x4*)&sB[0][r][seg * 32 + q * 8] = rb[q];
  }
  __syncthreads();

  for (int kb = 0; kb < nk; ++kb) {
    const int cur = kb & 1;
    // issue next stage's global loads early (latency hidden behind 16 WMMAs)
    if (kb + 1 < nk) {
      const u16* pa = gA + (size_t)(kb + 1) * BK;
      const u16* pb = gB + (size_t)(kb + 1) * BK;
#pragma unroll
      for (int q = 0; q < 4; ++q) {
        ra[q] = *(const f32x4*)(pa + q * 8);
        rb[q] = *(const f32x4*)(pb + q * 8);
      }
    }

    // B fragments for both 32-k steps (reused across all 4 M tiles)
    bf16x16 bfm[2][2];
#pragma unroll
    for (int ks = 0; ks < 2; ++ks) {
#pragma unroll
      for (int nt = 0; nt < 2; ++nt) {
        const u16* row = &sB[cur][wn * 32 + nt * 16 + l16][ks * 32];
        FragBF f;
        f.q[0] = *(const f32x4*)(row + hi * 16);       // K = hi*16 .. +7
        f.q[1] = *(const f32x4*)(row + hi * 16 + 8);   // K = hi*16+8 .. +15
        bfm[ks][nt] = f.bf;
      }
    }
    // A fragments per (M tile, k-step), each reused across the 2 N tiles
#pragma unroll
    for (int mt = 0; mt < 4; ++mt) {
#pragma unroll
      for (int ks = 0; ks < 2; ++ks) {
        const u16* row = &sA[cur][wm * 64 + mt * 16 + l16][ks * 32];
        FragBF f;
        f.q[0] = *(const f32x4*)(row + hi * 16);       // K = hi*16 .. +7
        f.q[1] = *(const f32x4*)(row + hi * 16 + 8);   // K = hi*16+8 .. +15
        bf16x16 af = f.bf;
#pragma unroll
        for (int nt = 0; nt < 2; ++nt)
          acc[mt][nt] = __builtin_amdgcn_wmma_f32_16x16x32_bf16(
              false, af, false, bfm[ks][nt], (short)0, acc[mt][nt], false, false);
      }
    }

    // deposit next stage while this one is still being consumed elsewhere
    if (kb + 1 < nk) {
      const int nxt = cur ^ 1;
#pragma unroll
      for (int q = 0; q < 4; ++q) {
        *(f32x4*)&sA[nxt][r][seg * 32 + q * 8] = ra[q];
        *(f32x4*)&sB[nxt][r][seg * 32 + q * 8] = rb[q];
      }
    }
    __syncthreads();
  }

  // epilogue: C/D layout -> lane l: n = l&15, m = vgpr + (l>>4)*8
#pragma unroll
  for (int mt = 0; mt < 4; ++mt) {
#pragma unroll
    for (int nt = 0; nt < 2; ++nt) {
      const int n  = bn + wn * 32 + nt * 16 + l16;
      const int mb = bm + wm * 64 + mt * 16 + hi * 8;
#pragma unroll
      for (int vv = 0; vv < 8; ++vv) {
        float val = acc[mt][nt][vv];
        float res;
        if (MODE == 0) {
          res = val;
        } else if (MODE == 1) {
          float warg = p0[n] + tanhf(val) * p1[n];
          float wdec = -log1pf(expf(-warg)) - 0.5f;
          res = expf(fmaxf(wdec, -10.0f));
        } else if (MODE == 2) {
          res = sigf(p0[n] + val * p1[n]);
        } else {
          res = p1[n] * sigf(val);
        }
        out[(size_t)(mb + vv) * Ndim + n] = res;
      }
    }
  }
}

// ---------------- post-GEMM fusion: v-mix, kk-normalize, k-gate ----------------
// one wave per (b,t,h) group of N=64 channels; lane owns channels lane and lane+32
__global__ __launch_bounds__(256) void post_kernel(
    float* __restrict__ K, float* __restrict__ V, float* __restrict__ KK /*in: v1raw*/,
    const float* __restrict__ A, const float* __restrict__ vfirst,
    const float* __restrict__ k_k, const float* __restrict__ k_a,
    const float* __restrict__ v0p, const float* __restrict__ v2p) {
  int tid = threadIdx.x, wave = tid >> 5, lane = tid & 31;
  int gid = blockIdx.x * 8 + wave;          // (b*T+t)*H + h
  int m = gid >> 4, h = gid & (Hq - 1);
  size_t base = (size_t)m * Cq + h * Nq;
  int cA = h * Nq + lane, cB = cA + 32;

  float kr0 = K[base + lane], kr1 = K[base + lane + 32];
  float kk0 = kr0 * k_k[cA], kk1 = kr1 * k_k[cB];
  float ss = kk0 * kk0 + kk1 * kk1;
#pragma unroll
  for (int msk = 16; msk >= 1; msk >>= 1) ss += __shfl_xor(ss, msk);
  float inv = 1.0f / fmaxf(sqrtf(ss), 1e-12f);

  float a0v = A[base + lane], a1v = A[base + lane + 32];
  float v10 = KK[base + lane], v11 = KK[base + lane + 32];    // v1raw (aliased)
  float vr0 = V[base + lane], vr1 = V[base + lane + 32];
  float vf0 = vfirst[base + lane], vf1 = vfirst[base + lane + 32];
  float sg0 = sigf(v0p[cA] + v10 * v2p[cA]);
  float sg1 = sigf(v0p[cB] + v11 * v2p[cB]);

  V[base + lane]      = vr0 + (vf0 - vr0) * sg0;
  V[base + lane + 32] = vr1 + (vf1 - vr1) * sg1;
  KK[base + lane]      = kk0 * inv;
  KK[base + lane + 32] = kk1 * inv;
  K[base + lane]      = kr0 * (1.0f + (a0v - 1.0f) * k_a[cA]);
  K[base + lane + 32] = kr1 * (1.0f + (a1v - 1.0f) * k_a[cB]);
}

// ---------------- sequential scan over T ----------------
// block per (b,h); thread owns row i = tid>>2, 16 columns at (tid&3)*16
__global__ __launch_bounds__(256) void scan_kernel(
    const float* __restrict__ Rp, const float* __restrict__ Wp,
    const float* __restrict__ Kp, const float* __restrict__ Vp,
    const float* __restrict__ KKp, const float* __restrict__ Ap,
    float* __restrict__ Yp) {
  const int bh = blockIdx.x;
  const int b = bh >> 4, h = bh & (Hq - 1);
  const int tid = threadIdx.x;
  const int i = tid >> 2;
  const int jq = (tid & 3) * 16;

  __shared__ float sh[2][6][Nq];   // 0:r 1:w 2:k 3:v 4:kk 5:a, double buffered
  float s[16];
#pragma unroll
  for (int e = 0; e < 16; ++e) s[e] = 0.0f;

  auto loadt = [&](int t, int bufi) {
    size_t off = ((size_t)(b * Tq + t)) * Cq + h * Nq;
    {
      int arr = tid >> 6, e = tid & 63;
      const float* p = (arr == 0) ? Rp : (arr == 1) ? Wp : (arr == 2) ? Kp : Vp;
      sh[bufi][arr][e] = p[off + e];
    }
    {
      int idx = tid + 256;
      int arr = idx >> 6, e = idx & 63;
      if (arr < 6) {
        const float* p = (arr == 4) ? KKp : Ap;
        sh[bufi][arr][e] = p[off + e];
      }
    }
  };

  int buf = 0;
  loadt(0, 0);
  __syncthreads();
  for (int t = 0; t < Tq; ++t) {
    if (t + 1 < Tq) loadt(t + 1, buf ^ 1);

    float w_i = sh[buf][1][i];
    float v_i = sh[buf][3][i];
    float sab = 0.0f;
#pragma unroll
    for (int e = 0; e < 16; ++e) sab += s[e] * sh[buf][4][jq + e];
    sab += __shfl_xor(sab, 1);
    sab += __shfl_xor(sab, 2);
    sab = -sab;                            // sum_k state[i,k] * (-kk[k])

    float yp = 0.0f;
#pragma unroll
    for (int e = 0; e < 16; ++e) {
      int j = jq + e;
      float kkv = sh[buf][4][j];
      float kav = kkv * sh[buf][5][j];     // kk*a
      s[e] = fmaf(s[e], w_i, fmaf(sab, kav, v_i * sh[buf][2][j]));
      yp = fmaf(s[e], sh[buf][0][j], yp);  // y uses updated state
    }
    yp += __shfl_xor(yp, 1);
    yp += __shfl_xor(yp, 2);
    if ((tid & 3) == 0)
      Yp[((size_t)(b * Tq + t)) * Cq + h * Nq + i] = yp;

    __syncthreads();
    buf ^= 1;
  }
}

// ---------------- GroupNorm + bonus + gate, bf16 out for Wo GEMM ----------------
__global__ __launch_bounds__(256) void gn_gate_kernel(
    const float* __restrict__ Y, const float* __restrict__ R,
    const float* __restrict__ K, const float* __restrict__ V,
    const float* __restrict__ G, const float* __restrict__ r_k,
    const float* __restrict__ ln_w, const float* __restrict__ ln_b,
    u16* __restrict__ YB) {
  int tid = threadIdx.x, wave = tid >> 5, lane = tid & 31;
  int gid = blockIdx.x * 8 + wave;
  int m = gid >> 4, h = gid & (Hq - 1);
  size_t base = (size_t)m * Cq + h * Nq;
  int cA = h * Nq + lane, cB = cA + 32;

  float y0 = Y[base + lane], y1 = Y[base + lane + 32];
  float r0 = R[base + lane], r1 = R[base + lane + 32];
  float k0 = K[base + lane], k1 = K[base + lane + 32];
  float v0v = V[base + lane], v1v = V[base + lane + 32];
  float g0 = G[base + lane], g1 = G[base + lane + 32];

  float sum = y0 + y1;
  float sq = y0 * y0 + y1 * y1;
  float dot = r0 * k0 * r_k[cA] + r1 * k1 * r_k[cB];
#pragma unroll
  for (int msk = 16; msk >= 1; msk >>= 1) {
    sum += __shfl_xor(sum, msk);
    sq  += __shfl_xor(sq, msk);
    dot += __shfl_xor(dot, msk);
  }
  float mean = sum * (1.0f / 64.0f);
  float var = sq * (1.0f / 64.0f) - mean * mean;
  float inv = rsqrtf(var + EPSgn);
  float o0 = ((y0 - mean) * inv * ln_w[cA] + ln_b[cA] + dot * v0v) * g0;
  float o1 = ((y1 - mean) * inv * ln_w[cB] + ln_b[cB] + dot * v1v) * g1;
  YB[base + lane]      = f32_to_bf16(o0);
  YB[base + lane + 32] = f32_to_bf16(o1);
}

// ---------------- host-side orchestration ----------------
extern "C" void kernel_launch(void* const* d_in, const int* in_sizes, int n_in,
                              void* d_out, int out_size, void* d_ws, size_t ws_size,
                              hipStream_t stream) {
  (void)in_sizes; (void)n_in; (void)out_size; (void)ws_size;
  const float* x      = (const float*)d_in[0];
  const float* vfirst = (const float*)d_in[1];
  const float* x_r = (const float*)d_in[2];
  const float* x_w = (const float*)d_in[3];
  const float* x_k = (const float*)d_in[4];
  const float* x_v = (const float*)d_in[5];
  const float* x_a = (const float*)d_in[6];
  const float* x_g = (const float*)d_in[7];
  const float* w0  = (const float*)d_in[8];
  const float* a0  = (const float*)d_in[9];
  const float* v0  = (const float*)d_in[10];
  const float* Wr  = (const float*)d_in[11];
  const float* Wk  = (const float*)d_in[12];
  const float* Wv  = (const float*)d_in[13];
  /* Wg = d_in[14] is unused in the forward pass */
  const float* Wo  = (const float*)d_in[15];
  const float* Ww1 = (const float*)d_in[16];
  const float* Wa1 = (const float*)d_in[17];
  const float* Wv1 = (const float*)d_in[18];
  const float* Wg1 = (const float*)d_in[19];
  const float* w2  = (const float*)d_in[20];
  const float* a2  = (const float*)d_in[21];
  const float* v2  = (const float*)d_in[22];
  const float* g2  = (const float*)d_in[23];
  const float* k_k = (const float*)d_in[24];
  const float* k_a = (const float*)d_in[25];
  const float* r_k = (const float*)d_in[26];
  const float* ln_w = (const float*)d_in[27];
  const float* ln_b = (const float*)d_in[28];

  const size_t CC = (size_t)Cq * Cq;     // 1M
  const size_t MC = (size_t)Mq * Cq;     // 4M
  // workspace layout (needs 176 MB):
  //   [0,16MB)  bf16 weights  (8 x CC)
  //   [16,64MB) bf16 mixes    (6 x MC)  -- consumed by GEMM phase
  //   [64,176MB) f32 r,w,k,v,kk(v1raw),a,g  (7 x MC)
  //   y(f32) aliases mix slots 0-1; yb(bf16) aliases mix slot 2
  u16* WB  = (u16*)d_ws;
  u16* MIX = WB + 8 * CC;
  float* F  = (float*)(MIX + 6 * MC);
  float* Rb = F + 0 * MC;
  float* Wd = F + 1 * MC;
  float* Kb = F + 2 * MC;
  float* Vb = F + 3 * MC;
  float* KKb = F + 4 * MC;   // first holds v1raw, then kk (normalized)
  float* Ab = F + 5 * MC;
  float* Gb = F + 6 * MC;
  float* Yf = (float*)MIX;          // 16MB alias over mix slots 0-1
  u16*   YB = MIX + 2 * MC;         // 8MB alias over mix slot 2

  // 1) weight f32 -> bf16
  const int cb = (int)(CC / 256);
  conv_f32_bf16<<<cb, 256, 0, stream>>>(Wr,  WB + 0 * CC, (int)CC);
  conv_f32_bf16<<<cb, 256, 0, stream>>>(Wk,  WB + 1 * CC, (int)CC);
  conv_f32_bf16<<<cb, 256, 0, stream>>>(Wv,  WB + 2 * CC, (int)CC);
  conv_f32_bf16<<<cb, 256, 0, stream>>>(Ww1, WB + 3 * CC, (int)CC);
  conv_f32_bf16<<<cb, 256, 0, stream>>>(Wa1, WB + 4 * CC, (int)CC);
  conv_f32_bf16<<<cb, 256, 0, stream>>>(Wv1, WB + 5 * CC, (int)CC);
  conv_f32_bf16<<<cb, 256, 0, stream>>>(Wg1, WB + 6 * CC, (int)CC);
  conv_f32_bf16<<<cb, 256, 0, stream>>>(Wo,  WB + 7 * CC, (int)CC);

  // 2) time-shift token mixing -> bf16 activations
  prep_mix<<<(int)(MC / 256), 256, 0, stream>>>(
      x, x_r, x_w, x_k, x_v, x_a, x_g,
      MIX + 0 * MC, MIX + 1 * MC, MIX + 2 * MC,
      MIX + 3 * MC, MIX + 4 * MC, MIX + 5 * MC);

  // 3) seven WMMA GEMMs with fused epilogues
  dim3 gg(Cq / BN, Mq / BM);
  gemm_bf16_wmma<0><<<gg, 256, 0, stream>>>(MIX + 0 * MC, WB + 0 * CC, Rb,  nullptr, nullptr, Mq, Cq, Cq);
  gemm_bf16_wmma<1><<<gg, 256, 0, stream>>>(MIX + 1 * MC, WB + 3 * CC, Wd,  w0, w2, Mq, Cq, Cq);
  gemm_bf16_wmma<0><<<gg, 256, 0, stream>>>(MIX + 2 * MC, WB + 1 * CC, Kb,  nullptr, nullptr, Mq, Cq, Cq);
  gemm_bf16_wmma<0><<<gg, 256, 0, stream>>>(MIX + 3 * MC, WB + 2 * CC, Vb,  nullptr, nullptr, Mq, Cq, Cq);
  gemm_bf16_wmma<0><<<gg, 256, 0, stream>>>(MIX + 3 * MC, WB + 5 * CC, KKb, nullptr, nullptr, Mq, Cq, Cq);
  gemm_bf16_wmma<2><<<gg, 256, 0, stream>>>(MIX + 4 * MC, WB + 4 * CC, Ab,  a0, a2, Mq, Cq, Cq);
  gemm_bf16_wmma<3><<<gg, 256, 0, stream>>>(MIX + 5 * MC, WB + 6 * CC, Gb,  nullptr, g2, Mq, Cq, Cq);

  // 4) fuse v-mixing, kk normalization, k gating
  post_kernel<<<Mq * Hq / 8, 256, 0, stream>>>(Kb, Vb, KKb, Ab, vfirst, k_k, k_a, v0, v2);

  // 5) sequential scan over T
  scan_kernel<<<Bq * Hq, 256, 0, stream>>>(Rb, Wd, Kb, Vb, KKb, Ab, Yf);

  // 6) GroupNorm + rk bonus + gate -> bf16
  gn_gate_kernel<<<Mq * Hq / 8, 256, 0, stream>>>(Yf, Rb, Kb, Vb, Gb, r_k, ln_w, ln_b, YB);

  // 7) output projection straight into d_out (f32)
  gemm_bf16_wmma<0><<<gg, 256, 0, stream>>>(YB, WB + 7 * CC, (float*)d_out, nullptr, nullptr, Mq, Cq, Cq);

  // 8) v_first passthrough (second output of the tuple)
  hipMemcpyAsync((float*)d_out + MC, vfirst, MC * sizeof(float),
                 hipMemcpyDeviceToDevice, stream);
}
